// MultiHeadEncDecAttention_8933531976131
// MI455X (gfx1250) — matrix operand order repaired and checked by reading, not compiled, not run on hardware
//
#include <hip/hip_runtime.h>
#include <hip/hip_bf16.h>
#include <cstdint>

// ---------------------------------------------------------------------------
// CDNA5 (gfx1250) fused multi-head attention:
//   Q = x@w_q+b_q ; K = enc@w_k+b_k ; V = enc@w_v+b_v   (bf16 WMMA GEMMs)
//   flash-attention per head (bf16 WMMA, fp32 online softmax,
//                             double-buffered GLOBAL_LOAD_ASYNC_TO_LDS tiles)
//   out = attn@w_o+b_o                                  (bf16 WMMA GEMM)
// ---------------------------------------------------------------------------

typedef __attribute__((ext_vector_type(16))) __bf16 v16bf;
typedef __attribute__((ext_vector_type(8)))  float  v8f;

union FragA { v16bf v; uint32_t u[8]; };

__device__ __forceinline__ v8f wmma_bf16(v16bf a, v16bf b, v8f c) {
    return __builtin_amdgcn_wmma_f32_16x16x32_bf16(
        /*neg_a=*/false, a, /*neg_b=*/false, b,
        /*c_mod=*/(short)0, c, /*reuse_a=*/false, /*reuse_b=*/false);
}

__device__ __forceinline__ unsigned short bf16_bits(float f) {
    uint32_t u = __builtin_bit_cast(uint32_t, f);
    u = (u + 0x7FFFu + ((u >> 16) & 1u)) >> 16;   // RNE
    return (unsigned short)u;
}
__device__ __forceinline__ uint32_t pack_bf16x2(float lo, float hi) {
    return (uint32_t)bf16_bits(lo) | ((uint32_t)bf16_bits(hi) << 16);
}

// ISA 16-bit A-matrix 16x32 layout: element-pair p (0..7) of lane-half g maps
// to K = g*8+2p (p<4) or 16+g*8+2(p-4) (p>=4).  Pairs are consecutive K.
__device__ __forceinline__ int pairK(int p, int g) {
    return (p < 4) ? (g * 8 + 2 * p) : (16 + g * 8 + 2 * (p - 4));
}

__device__ __forceinline__ void store_val(unsigned short* C, size_t i, float v) { C[i] = bf16_bits(v); }
__device__ __forceinline__ void store_val(float* C, size_t i, float v)          { C[i] = v; }

// ---- CDNA5 async copy global -> LDS (ASYNCcnt-tracked, no VGPR round trip)
__device__ __forceinline__ void async_b128(uint32_t ldsByte, const uint32_t* g) {
    asm volatile("global_load_async_to_lds_b128 %0, %1, off"
                 :: "v"(ldsByte), "v"(g) : "memory");
}
__device__ __forceinline__ void wait_async0() {
    asm volatile("s_wait_asynccnt 0x0" ::: "memory");
}
// LDS byte address: low 32 bits of the generic address
__device__ __forceinline__ uint32_t lds_byte_addr(const void* p) {
    return (uint32_t)(uintptr_t)p;
}

// ---------------------------------------------------------------------------
// GEMM:  C[M x N] = A[M x K] * W[K x N] + bias[N]
// 128x64 tile, BK=32, 256 threads (8 waves), each wave -> 32x32 (4 WMMA frags)
// ---------------------------------------------------------------------------
template <typename OutT, bool TRANS>
__global__ __launch_bounds__(256)
void gemm_bias_kernel(const float* __restrict__ A, const float* __restrict__ W,
                      const float* __restrict__ bias, OutT* __restrict__ C,
                      int M, int K, int N) {
    __shared__ uint32_t lds_a[128 * 16];   // [m][k/2] packed bf16 pairs
    __shared__ uint32_t lds_b[64 * 16];    // [n][k/2] packed bf16 pairs (K-major pairs)
    const int t    = threadIdx.x;
    const int lane = t & 31;
    const int wave = t >> 5;
    const int half = lane >> 4;
    const int l16  = lane & 15;
    const int row0 = blockIdx.y * 128;
    const int col0 = blockIdx.x * 64;
    const int wm   = (wave >> 1) * 32;
    const int wn   = (wave & 1) * 32;

    v8f acc[2][2] = {};

    for (int kt = 0; kt < K; kt += 32) {
        // ---- stage A tile (128x32 fp32 -> bf16 pairs), coalesced float2 loads
#pragma unroll
        for (int i = 0; i < 8; ++i) {
            int q = i * 256 + t;
            int m = q >> 4, kp = q & 15;
            const float* src = A + (size_t)(row0 + m) * K + kt + 2 * kp;
            float2 f = *(const float2*)src;
            lds_a[m * 16 + kp] = pack_bf16x2(f.x, f.y);
        }
        // ---- stage W tile (32x64), pair = two consecutive K rows of one column
#pragma unroll
        for (int i = 0; i < 4; ++i) {
            int q = i * 256 + t;
            int n = q & 63, kp = q >> 6;
            const float* src = W + (size_t)(kt + 2 * kp) * N + col0 + n;
            lds_b[n * 16 + kp] = pack_bf16x2(src[0], src[N]);
        }
        // ---- warm L2/WGP$ for next BK tile (global_prefetch_b8)
        if (kt + 32 < K) {
            __builtin_prefetch(A + (size_t)(row0 + (t >> 1)) * K + kt + 32 + (t & 1) * 16, 0, 0);
            __builtin_prefetch(W + (size_t)(kt + 32 + (t >> 3)) * N + col0 + (t & 7) * 8, 0, 0);
        }
        __syncthreads();

        FragA a[2], b[2];
#pragma unroll
        for (int mf = 0; mf < 2; ++mf)
#pragma unroll
            for (int p = 0; p < 8; ++p)
                a[mf].u[p] = lds_a[(wm + mf * 16 + l16) * 16 + (pairK(p, half) >> 1)];
#pragma unroll
        for (int nf = 0; nf < 2; ++nf)
#pragma unroll
            for (int p = 0; p < 8; ++p)
                b[nf].u[p] = lds_b[(wn + nf * 16 + l16) * 16 + (pairK(p, half) >> 1)];
#pragma unroll
        for (int mf = 0; mf < 2; ++mf)
#pragma unroll
            for (int nf = 0; nf < 2; ++nf)
                acc[mf][nf] = wmma_bf16(a[mf].v, b[nf].v, acc[mf][nf]);
        __syncthreads();
    }

    // ---- epilogue: D layout (lane-half picks rows r / r+8), add bias, store
#pragma unroll
    for (int mf = 0; mf < 2; ++mf)
#pragma unroll
        for (int nf = 0; nf < 2; ++nf)
#pragma unroll
            for (int r = 0; r < 8; ++r) {
                int row = row0 + wm + mf * 16 + r + half * 8;
                int col = col0 + wn + nf * 16 + l16;
                float v = acc[mf][nf][r] + bias[col];
                size_t idx = TRANS ? ((size_t)col * M + row) : ((size_t)row * N + col);
                store_val(C, idx, v);
            }
}

// ---------------------------------------------------------------------------
// Flash attention: block = 64 query rows x 1 head, 128 threads (4 waves),
// each wave owns 16 query rows.  Q/K bf16 [4096][768]; V bf16 transposed
// [768][4096].  K/V tiles double-buffered via GLOBAL_LOAD_ASYNC_TO_LDS_B128.
// Online softmax in fp32.  Output fp32 [4096][768].
// ---------------------------------------------------------------------------
__global__ __launch_bounds__(128)
void attention_kernel(const uint32_t* __restrict__ Q, const uint32_t* __restrict__ Km,
                      const uint32_t* __restrict__ Vt, float* __restrict__ Out) {
    __shared__ uint32_t       lds_k[2][64 * 32];    // [buf][key][d/2]
    __shared__ uint32_t       lds_v[2][64 * 32];    // [buf][d][key/2]
    __shared__ unsigned short lds_p[4 * 16 * 64];   // per-wave P scratch

    const int t = threadIdx.x, lane = t & 31, wave = t >> 5;
    const int half = lane >> 4, l16 = lane & 15;
    const int qt = blockIdx.x, h = blockIdx.y;
    const int hb = h * 64;

    // Issue the async copies for one 64-key tile (4 x b128 per thread per array)
    auto stage = [&](int kt, int buf) {
#pragma unroll
        for (int i = 0; i < 4; ++i) {
            int q = (i * 128 + t) * 4;              // dword index, 16B granules
            int key = q >> 5, kp = q & 31;
            async_b128(lds_byte_addr(&lds_k[buf][q]),
                       Km + ((uint32_t)((kt * 64 + key) * 768 + hb) >> 1) + kp);
        }
#pragma unroll
        for (int i = 0; i < 4; ++i) {
            int q = (i * 128 + t) * 4;
            int d = q >> 5, kp = q & 31;
            async_b128(lds_byte_addr(&lds_v[buf][q]),
                       Vt + ((uint32_t)((hb + d) * 4096 + kt * 64) >> 1) + kp);
        }
    };

    // Q fragments for this wave's 16 rows (kept in registers for whole loop)
    FragA qa[2];
    {
        const int qrow = qt * 64 + wave * 16 + l16;
        const uint32_t base = (uint32_t)(qrow * 768 + hb) >> 1;
#pragma unroll
        for (int c = 0; c < 2; ++c)
#pragma unroll
            for (int p = 0; p < 8; ++p)
                qa[c].u[p] = Q[base + c * 16 + (pairK(p, half) >> 1)];
    }

    v8f acc[4] = {};
    float m_run[8], l_run[8];
#pragma unroll
    for (int r = 0; r < 8; ++r) { m_run[r] = -1e30f; l_run[r] = 0.f; }

    stage(0, 0);
    wait_async0();
    __syncthreads();

    for (int kt = 0; kt < 64; ++kt) {
        const int cur = kt & 1;
        const uint32_t* ldsK = lds_k[cur];
        const uint32_t* ldsV = lds_v[cur];

        // ---- prefetch next tile into the other buffer while we compute
        if (kt + 1 < 64) stage(kt + 1, cur ^ 1);

        // ---- S = Q * K^T  (4 key-frags x 2 d-chunks = 8 WMMAs)
        v8f s[4];
#pragma unroll
        for (int f = 0; f < 4; ++f) {
            v8f z = {};
#pragma unroll
            for (int c = 0; c < 2; ++c) {
                FragA kb;
#pragma unroll
                for (int p = 0; p < 8; ++p)
                    kb.u[p] = ldsK[(f * 16 + l16) * 32 + c * 16 + (pairK(p, half) >> 1)];
                z = wmma_bf16(qa[c].v, kb.v, z);
            }
            s[f] = z;
        }

        // ---- online softmax (row r+8*half lives in VGPR r across 16 lanes)
#pragma unroll
        for (int r = 0; r < 8; ++r) {
            float mx = -1e30f;
#pragma unroll
            for (int f = 0; f < 4; ++f) { s[f][r] *= 0.125f; mx = fmaxf(mx, s[f][r]); }
#pragma unroll
            for (int o = 1; o < 16; o <<= 1) mx = fmaxf(mx, __shfl_xor(mx, o, 32));
            float m_new = fmaxf(m_run[r], mx);
            float fac   = __expf(m_run[r] - m_new);
            float sum   = 0.f;
#pragma unroll
            for (int f = 0; f < 4; ++f) { float e = __expf(s[f][r] - m_new); s[f][r] = e; sum += e; }
#pragma unroll
            for (int o = 1; o < 16; o <<= 1) sum += __shfl_xor(sum, o, 32);
            l_run[r] = l_run[r] * fac + sum;
            m_run[r] = m_new;
#pragma unroll
            for (int df = 0; df < 4; ++df) acc[df][r] *= fac;
        }

        // ---- re-layout P: D-layout -> A-layout via per-wave LDS scratch
        unsigned short* pp = lds_p + wave * 1024;
#pragma unroll
        for (int f = 0; f < 4; ++f)
#pragma unroll
            for (int r = 0; r < 8; ++r)
                pp[(r + half * 8) * 64 + f * 16 + l16] = bf16_bits(s[f][r]);
        asm volatile("s_wait_dscnt 0x0" ::: "memory");

        const uint32_t* pu = (const uint32_t*)lds_p + wave * 512;
        FragA pa[2];
#pragma unroll
        for (int c = 0; c < 2; ++c)
#pragma unroll
            for (int p = 0; p < 8; ++p)
                pa[c].u[p] = pu[l16 * 32 + c * 16 + (pairK(p, half) >> 1)];

        // ---- O += P * V  (2 key-chunks x 4 d-frags = 8 WMMAs)
#pragma unroll
        for (int c = 0; c < 2; ++c)
#pragma unroll
            for (int df = 0; df < 4; ++df) {
                FragA vb;
#pragma unroll
                for (int p = 0; p < 8; ++p)
                    vb.u[p] = ldsV[(df * 16 + l16) * 32 + c * 16 + (pairK(p, half) >> 1)];
                acc[df] = wmma_bf16(pa[c].v, vb.v, acc[df]);
            }

        // ---- my async copies done, then block-wide visibility for next tile
        wait_async0();
        __syncthreads();
    }

    // ---- epilogue: normalize and write fp32
#pragma unroll
    for (int df = 0; df < 4; ++df)
#pragma unroll
        for (int r = 0; r < 8; ++r) {
            int row = qt * 64 + wave * 16 + r + half * 8;
            int col = hb + df * 16 + l16;
            Out[(size_t)row * 768 + col] = acc[df][r] / l_run[r];
        }
}

// ---------------------------------------------------------------------------
extern "C" void kernel_launch(void* const* d_in, const int* in_sizes, int n_in,
                              void* d_out, int out_size, void* d_ws, size_t ws_size,
                              hipStream_t stream) {
    const float* x   = (const float*)d_in[0];
    const float* enc = (const float*)d_in[1];
    const float* w_q = (const float*)d_in[2];
    const float* b_q = (const float*)d_in[3];
    const float* w_k = (const float*)d_in[4];
    const float* b_k = (const float*)d_in[5];
    const float* w_v = (const float*)d_in[6];
    const float* b_v = (const float*)d_in[7];
    const float* w_o = (const float*)d_in[8];
    const float* b_o = (const float*)d_in[9];
    float* out = (float*)d_out;

    const size_t NT = (size_t)4096 * 768;
    unsigned short* Qb = (unsigned short*)d_ws;     // bf16 [4096][768]
    unsigned short* Kb = Qb + NT;                   // bf16 [4096][768]
    unsigned short* Vt = Kb + NT;                   // bf16 [768][4096] (transposed)
    float*        attn = (float*)(Vt + NT);         // f32  [4096][768]

    dim3 gg(12, 32), gb(256);
    gemm_bias_kernel<unsigned short, false><<<gg, gb, 0, stream>>>(x,   w_q, b_q, Qb, 4096, 768, 768);
    gemm_bias_kernel<unsigned short, false><<<gg, gb, 0, stream>>>(enc, w_k, b_k, Kb, 4096, 768, 768);
    gemm_bias_kernel<unsigned short, true ><<<gg, gb, 0, stream>>>(enc, w_v, b_v, Vt, 4096, 768, 768);
    attention_kernel<<<dim3(64, 12), 128, 0, stream>>>(
        (const uint32_t*)Qb, (const uint32_t*)Kb, (const uint32_t*)Vt, attn);
    gemm_bias_kernel<float, false><<<gg, gb, 0, stream>>>(attn, w_o, b_o, out, 4096, 768, 768);
}
